// EMAVectorQuantizer_12584254177748
// MI455X (gfx1250) — matrix operand order repaired and checked by reading, not compiled, not run on hardware
//
#include <hip/hip_runtime.h>
#include <hip/hip_bf16.h>

typedef __attribute__((ext_vector_type(16))) _Float16 v16h;
typedef __attribute__((ext_vector_type(8)))  float    v8f;
typedef int v4i __attribute__((vector_size(4 * sizeof(int))));

#define AS1 __attribute__((address_space(1)))
#define AS3 __attribute__((address_space(3)))

#define N_ROWS   32768          // 32*256*32*32 / 256
#define DIM      256            // EMBED_DIM
#define K_CODES  1024           // N_EMBED
#define N_ELEM   (N_ROWS * DIM) // 8388608 elements of quantized output

#define LDS_ROW  264            // 256 halves + 8 pad (528B row -> bank spread)
#define ROWS_PER_BLOCK 256      // 8 waves * 32 rows

// workspace layout (byte offsets into d_ws)
#define WS_LOSS   0
#define WS_WNORM  1024
#define WS_WH     8192
#define WS_IDX    (8192 + K_CODES * DIM * 2)   // after 512KB of f16 codebook

#if __has_builtin(__builtin_amdgcn_global_load_async_to_lds_b128)
#define USE_ASYNC 1
#else
#define USE_ASYNC 0
#endif

// ---------------------------------------------------------------------------
// Kernel 1: convert codebook fp32 -> f16, compute ||e_k||^2, zero loss acc.
// ---------------------------------------------------------------------------
__global__ __launch_bounds__(256)
void vq_prep(const float* __restrict__ weight, float* __restrict__ loss_acc,
             float* __restrict__ wnorm, _Float16* __restrict__ wh) {
    const int k = blockIdx.x;
    const int d = threadIdx.x;
    const float w = weight[k * DIM + d];
    wh[k * DIM + d] = (_Float16)w;

    __shared__ float red[256];
    red[d] = w * w;
    __syncthreads();
    for (int s = 128; s > 0; s >>= 1) {
        if (d < s) red[d] += red[d + s];
        __syncthreads();
    }
    if (d == 0) wnorm[k] = red[0];
    if (k == 0 && d == 0) *loss_acc = 0.0f;
}

// ---------------------------------------------------------------------------
// Stage one 16-code B tile (8KB) into LDS; 256 threads x 2 x 16B chunks.
// Async DMA path (ASYNCcnt-tracked) when available; sync fallback otherwise.
// ---------------------------------------------------------------------------
__device__ __forceinline__
void stage_tile(const _Float16* __restrict__ wh, _Float16* lds_buf,
                int tile, int tid) {
#pragma unroll
    for (int h = 0; h < 2; ++h) {
        const int q = tid + h * 256;       // chunk id 0..511
        const int c = q >> 5;              // code row within tile (0..15)
        const int j = q & 31;              // 16B chunk within row (0..31)
        char* gsrc = (char*)wh + (size_t)(tile * 16 + c) * (DIM * 2) + j * 16;
        char* ldst = (char*)lds_buf + c * (LDS_ROW * 2) + j * 16;
#if USE_ASYNC
        __builtin_amdgcn_global_load_async_to_lds_b128(
            (AS1 v4i*)gsrc, (AS3 v4i*)ldst, 0, 0);
#else
        *(uint4*)ldst = *(const uint4*)gsrc;
#endif
    }
}

__device__ __forceinline__ void wait_async_done() {
#if USE_ASYNC
#if __has_builtin(__builtin_amdgcn_s_wait_asynccnt)
    __builtin_amdgcn_s_wait_asynccnt(0);
#else
    asm volatile("s_wait_asynccnt 0x0" ::: "memory");
#endif
#endif
}

// ---------------------------------------------------------------------------
// Kernel 2: fused GEMM (f16 WMMA) + running argmin over the 1024 codes.
// 8 waves/block share double-buffered LDS B tiles (async-staged).
// Each wave owns M=32 rows (two A fragments / two independent accumulator
// chains) so every staged B tile feeds 16 WMMAs: LDS-read traffic per FLOP
// is halved and the matrix pipe sees two interleaved dependency chains.
// dist(n,k) = ||e_k||^2 - 2 * x_n . e_k   (||x_n||^2 dropped: row-constant)
// ---------------------------------------------------------------------------
__global__ __launch_bounds__(256)
void vq_argmin(const float* __restrict__ x, const _Float16* __restrict__ wh,
               const float* __restrict__ wnorm, int* __restrict__ idx_ws,
               float* __restrict__ out_idx) {
    __shared__ __align__(16) _Float16 Bs[2][16 * LDS_ROW];

    const int tid  = threadIdx.x;
    const int lane = tid & 31;
    const int wave = tid >> 5;
    const int g    = lane >> 4;      // half-wave group (0 or 1)
    const int lm   = lane & 15;
    const int base = blockIdx.x * ROWS_PER_BLOCK + wave * 32;
    const int r0   = base + lm;      // A-fragment-0 row owned by this lane
    const int r1   = base + 16 + lm; // A-fragment-1 row owned by this lane

    // --- Load A fragments: 2 x (16 rows x 256), fp32 -> f16, CDNA5 layout ---
    // 16-bit A 16x32: lanes 0-15 hold K chunks [0..7],[16..23];
    //                 lanes 16-31 hold K chunks [8..15],[24..31].
    v16h a0[8], a1[8];
#pragma unroll
    for (int kk = 0; kk < 8; ++kk) {
        const float* p0 = x + (size_t)r0 * DIM + kk * 32 + g * 8;
        const float* p1 = x + (size_t)r1 * DIM + kk * 32 + g * 8;
#pragma unroll
        for (int e = 0; e < 8; ++e) a0[kk][e]     = (_Float16)p0[e];
#pragma unroll
        for (int e = 0; e < 8; ++e) a0[kk][8 + e] = (_Float16)p0[16 + e];
#pragma unroll
        for (int e = 0; e < 8; ++e) a1[kk][e]     = (_Float16)p1[e];
#pragma unroll
        for (int e = 0; e < 8; ++e) a1[kk][8 + e] = (_Float16)p1[16 + e];
    }

    float minv0[8], minv1[8];
    int   mini0[8], mini1[8];
#pragma unroll
    for (int j = 0; j < 8; ++j) {
        minv0[j] = 3.0e38f; mini0[j] = 0;
        minv1[j] = 3.0e38f; mini1[j] = 0;
    }

    stage_tile(wh, Bs[0], 0, tid);                 // prefetch tile 0

    for (int t = 0; t < K_CODES / 16; ++t) {
        const int cur = t & 1;
        wait_async_done();           // this wave's tile-t transfers complete
        __syncthreads();             // all waves' tile-t transfers complete

        if (t + 1 < K_CODES / 16)    // DMA tile t+1 while computing tile t
            stage_tile(wh, Bs[cur ^ 1], t + 1, tid);

        const int kcode = t * 16 + lm;   // B column owned by this lane
        v8f c0 = {}, c1 = {};
        // B layout (32x16 f16): lane holds column n=lane%16, K chunks
        // [8g..8g+7] and [16+8g..16+8g+7] (matches A).
        const _Float16* bp = &Bs[cur][lm * LDS_ROW + g * 8];
#pragma unroll
        for (int kk = 0; kk < 8; ++kk) {
            v16h b;
            { uint4 lo = *(const uint4*)(bp + kk * 32); uint4 hi = *(const uint4*)(bp + kk * 32 + 16);
              __builtin_memcpy(&b, &lo, 16); __builtin_memcpy((char*)&b + 16, &hi, 16); }
            c0 = __builtin_amdgcn_wmma_f32_16x16x32_f16(
                     false, a0[kk], false, b, (short)0, c0, false, false);
            c1 = __builtin_amdgcn_wmma_f32_16x16x32_f16(
                     false, a1[kk], false, b, (short)0, c1, false, false);
        }
        const float wn = wnorm[kcode];
        // C layout: VGPR j of this lane = (M = j + 8*g, N = lm)
#pragma unroll
        for (int j = 0; j < 8; ++j) {
            const float d0 = wn - 2.0f * c0[j];
            if (d0 < minv0[j]) { minv0[j] = d0; mini0[j] = kcode; }
            const float d1 = wn - 2.0f * c1[j];
            if (d1 < minv1[j]) { minv1[j] = d1; mini1[j] = kcode; }
        }
        __syncthreads();             // done reading Bs[cur] before reuse
    }

    // --- Reduce argmin across the 16 lanes of each half-wave ---
#pragma unroll
    for (int s = 1; s < 16; s <<= 1) {
#pragma unroll
        for (int j = 0; j < 8; ++j) {
            float ov = __shfl_xor(minv0[j], s, 32);
            int   oi = __shfl_xor(mini0[j], s, 32);
            if (ov < minv0[j] || (ov == minv0[j] && oi < mini0[j])) {
                minv0[j] = ov; mini0[j] = oi;
            }
            ov = __shfl_xor(minv1[j], s, 32);
            oi = __shfl_xor(mini1[j], s, 32);
            if (ov < minv1[j] || (ov == minv1[j] && oi < mini1[j])) {
                minv1[j] = ov; mini1[j] = oi;
            }
        }
    }

    if (lm == 0) {
#pragma unroll
        for (int j = 0; j < 8; ++j) {
            const int row0 = base + j + 8 * g;
            idx_ws[row0]  = mini0[j];
            out_idx[row0] = (float)mini0[j];
            const int row1 = base + 16 + j + 8 * g;
            idx_ws[row1]  = mini1[j];
            out_idx[row1] = (float)mini1[j];
        }
    }
}

// ---------------------------------------------------------------------------
// Kernel 3: gather fp32 codebook rows -> quantized output, fused MSE partial
// sum (block reduce + one float atomic per block).
// ---------------------------------------------------------------------------
__global__ __launch_bounds__(256)
void vq_gather_loss(const float* __restrict__ x, const float* __restrict__ weight,
                    const int* __restrict__ idx_ws, float* __restrict__ outq,
                    float* __restrict__ loss_acc) {
    float acc = 0.0f;
    const int nvec = N_ELEM / 4;                  // float4 granules
    for (int i = blockIdx.x * blockDim.x + threadIdx.x; i < nvec;
         i += gridDim.x * blockDim.x) {
        const int n  = i >> 6;                    // 64 float4 per row
        const int d4 = i & 63;
        const int k  = idx_ws[n];
        const float4 q   = ((const float4*)weight)[k * 64 + d4];
        const float4 xin = ((const float4*)x)[i];
        ((float4*)outq)[i] = q;
        const float dx = q.x - xin.x, dy = q.y - xin.y;
        const float dz = q.z - xin.z, dw = q.w - xin.w;
        acc += dx * dx + dy * dy + dz * dz + dw * dw;
    }
    __shared__ float red[256];
    red[threadIdx.x] = acc;
    __syncthreads();
    for (int s = 128; s > 0; s >>= 1) {
        if (threadIdx.x < s) red[threadIdx.x] += red[threadIdx.x + s];
        __syncthreads();
    }
    if (threadIdx.x == 0) atomicAdd(loss_acc, red[0]);
}

// ---------------------------------------------------------------------------
// Kernel 4: loss = (q_latent + BETA*e_latent) = 1.25 * sum / N_ELEM
// ---------------------------------------------------------------------------
__global__ void vq_finalize(const float* __restrict__ loss_acc,
                            float* __restrict__ out_loss) {
    *out_loss = *loss_acc * (1.25f / (float)N_ELEM);
}

// ---------------------------------------------------------------------------
extern "C" void kernel_launch(void* const* d_in, const int* in_sizes, int n_in,
                              void* d_out, int out_size, void* d_ws, size_t ws_size,
                              hipStream_t stream) {
    const float* x = (const float*)d_in[0];   // [32,256,32,32] fp32
    const float* w = (const float*)d_in[1];   // [1024,256]    fp32
    float* out = (float*)d_out;               // quantized | loss | idx(as float)

    char* ws = (char*)d_ws;
    float*    loss_acc = (float*)(ws + WS_LOSS);
    float*    wnorm    = (float*)(ws + WS_WNORM);
    _Float16* wh       = (_Float16*)(ws + WS_WH);
    int*      idx_ws   = (int*)(ws + WS_IDX);

    vq_prep<<<K_CODES, 256, 0, stream>>>(w, loss_acc, wnorm, wh);
    vq_argmin<<<N_ROWS / ROWS_PER_BLOCK, 256, 0, stream>>>(x, wh, wnorm, idx_ws,
                                                           out + N_ELEM + 1);
    vq_gather_loss<<<2048, 256, 0, stream>>>(x, w, idx_ws, out, loss_acc);
    vq_finalize<<<1, 1, 0, stream>>>(loss_acc, out + N_ELEM);
}
